// CGEncoder_25374666785537
// MI455X (gfx1250) — matrix operand order repaired and checked by reading, not compiled, not run on hardware
//
#include <hip/hip_runtime.h>
#include <hip/hip_bf16.h>
#include <math.h>

#define NA      30000
#define NE      120000
#define NCG     1500
#define FEAT    128
#define F3      384
#define NRBF    20
#define NCONV   3
#define CUTF    5.0f
#define PI_F    3.14159265358979323846f
#define NAF     (NA * FEAT)                    // one SoA vector plane

typedef __attribute__((ext_vector_type(16))) __bf16 v16bf;
typedef __attribute__((ext_vector_type(8)))  float  v8f;

__device__ __forceinline__ float silu_fast(float x)
{
    return x * __builtin_amdgcn_rcpf(1.0f + __expf(-x));
}

// ---------------------------------------------------------------------------
// Tiled GEMM: C[M,N] = act(A[M,K] @ B[K,N] + bias), all unit column strides.
//   block = 256 threads = 8 waves = 8 M-tiles x one 64-column group
//   B panel (K x 64) staged once in LDS as bf16, transposed, stride padded to
//   KP = K+16 halves so every v16bf fragment read is one 32B-aligned LDS load
//   and is shared by all 8 waves.  Each wave keeps 4 accumulators (16x64
//   output) -> A frag + f32->bf16 cvt reused 4x.  A frag loaded as float4
//   (16B aligned: lda in {128,256,384}, k-offsets multiples of 8).
//   K is a template constant (128 / 256) -> K-loop fully unrolled.
// ---------------------------------------------------------------------------
template<int K>
__global__ __launch_bounds__(256)
void gemm_wmma_bf16(const float* __restrict__ A, int lda,
                    const float* __restrict__ B, int ldb,
                    float* __restrict__ C, int ldc,
                    const float* __restrict__ bias, int act, int M)
{
    constexpr int KP = K + 16;                 // padded stride (halves), 32B units
    __shared__ __bf16 Bt[64 * KP];

    const int tid = threadIdx.x;
    const int n0  = blockIdx.y << 6;           // first column of this 64-group

    // cooperative fill: global B[K x 64] (f32) -> LDS Bt[col][k] (bf16)
#pragma unroll 4
    for (int it = 0; it < (K * 64) / 256; ++it) {
        int idx = it * 256 + tid;
        int col = idx & 63;                    // consecutive tid -> coalesced cols
        int k   = idx >> 6;
        Bt[col * KP + k] = (__bf16)B[(long)k * ldb + n0 + col];
    }
    __syncthreads();                           // only barrier: before any exit

    const int mtile = blockIdx.x * 8 + (tid >> 5);
    if (mtile >= (M >> 4)) return;             // wave-uniform; EXEC stays full

    const int lane = tid & 31;
    const int half = lane >> 4;                // 0: lanes 0-15, 1: lanes 16-31
    const int l15  = lane & 15;
    const int m    = (mtile << 4) + l15;       // A row this lane feeds
    const int ka0  = half ? 16 : 0;            // 16-bit A frag half-wave K base
    const int kb0  = half ? 16 : 0;            // 16-bit B frag half-wave K base

    const v8f zero = {0.f, 0.f, 0.f, 0.f, 0.f, 0.f, 0.f, 0.f};
    v8f acc[4] = {zero, zero, zero, zero};
    const float* Arow = A + (long)m * lda;

#pragma unroll
    for (int kc = 0; kc < K; kc += 32) {
        // A 16x32 bf16 frag: halves 0-15 <- K=ka0+0..15
        const float4* Af = (const float4*)(Arow + kc + ka0);
        float4 q0 = Af[0], q1 = Af[1], q2 = Af[2], q3 = Af[3];
        v16bf a;
        a[0]  = (__bf16)q0.x; a[1]  = (__bf16)q0.y; a[2]  = (__bf16)q0.z; a[3]  = (__bf16)q0.w;
        a[4]  = (__bf16)q1.x; a[5]  = (__bf16)q1.y; a[6]  = (__bf16)q1.z; a[7]  = (__bf16)q1.w;
        a[8]  = (__bf16)q2.x; a[9]  = (__bf16)q2.y; a[10] = (__bf16)q2.z; a[11] = (__bf16)q2.w;
        a[12] = (__bf16)q3.x; a[13] = (__bf16)q3.y; a[14] = (__bf16)q3.z; a[15] = (__bf16)q3.w;

        const __bf16* bp = Bt + kc + kb0;      // 32B-aligned (kc%32==0, kb0 in {0,16})
#pragma unroll
        for (int t = 0; t < 4; ++t) {
            v16bf b = *(const v16bf*)(bp + (l15 + (t << 4)) * KP);
            acc[t] = __builtin_amdgcn_wmma_f32_16x16x32_bf16(false, a, false, b,
                                                             (short)0, acc[t],
                                                             false, false);
        }
    }

    const int mr0 = (mtile << 4) + (half ? 8 : 0);  // C frag: elem r -> M = mr0+r
#pragma unroll
    for (int t = 0; t < 4; ++t) {
        const int   n  = n0 + (t << 4) + l15;
        const float bv = bias ? bias[n] : 0.0f;
#pragma unroll
        for (int r = 0; r < 8; ++r) {
            float x = acc[t][r] + bv;
            if (act) x = silu_fast(x);
            C[(long)(mr0 + r) * ldc + n] = x;
        }
    }
}

// ---------------------------------------------------------------------------
// Elementwise / scatter kernels (v / uv / vv / dv are SoA: [3][NA,128])
// ---------------------------------------------------------------------------
__global__ void k_embed(const int* __restrict__ z, const float* __restrict__ tab,
                        float* __restrict__ s)
{
    int idx = blockIdx.x * blockDim.x + threadIdx.x;
    if (idx >= NAF) return;
    int n = idx >> 7, f = idx & 127;
    s[idx] = tab[z[n] * FEAT + f];
}

__global__ void k_count(const int* __restrict__ mapping, float* __restrict__ cnt)
{
    int n = blockIdx.x * blockDim.x + threadIdx.x;
    if (n >= NA) return;
    atomicAdd(cnt + mapping[n], 1.0f);
}

// fused per-edge message: RBF(K=20) embed + gather phi[j]/v[j] + atomic scatter
__global__ void k_edge_msg(const float* __restrict__ xyz, const int* __restrict__ nbr,
                           const float* __restrict__ phi, const float* __restrict__ v,
                           const float* __restrict__ dw, const float* __restrict__ db,
                           float* __restrict__ ds, float* __restrict__ dv)
{
    const int e = blockIdx.x;                 // directed edge id, [0, 2*NE)
    const int f = threadIdx.x;                // feature, [0,128)
    int i0, j;
    if (e < NE) { i0 = nbr[2 * e];     j = nbr[2 * e + 1]; }
    else        { int u = e - NE; i0 = nbr[2 * u + 1]; j = nbr[2 * u]; }

    __shared__ float sh_rbf[NRBF];
    __shared__ float sh_unit[3];
    __shared__ float sh_env;
    if (f < NRBF) {
        float rx = xyz[3 * j + 0] - xyz[3 * i0 + 0];
        float ry = xyz[3 * j + 1] - xyz[3 * i0 + 1];
        float rz = xyz[3 * j + 2] - xyz[3 * i0 + 2];
        float d  = sqrtf(rx * rx + ry * ry + rz * rz + 3e-15f);
        sh_rbf[f] = __sinf((float)(f + 1) * (PI_F / CUTF) * d) * __builtin_amdgcn_rcpf(d);
        if (f == 0) {
            float id = __builtin_amdgcn_rcpf(d);
            sh_unit[0] = rx * id; sh_unit[1] = ry * id; sh_unit[2] = rz * id;
            sh_env = (d < CUTF) ? 0.5f * (__cosf(PI_F * d / CUTF) + 1.0f) : 0.0f;
        }
    }
    __syncthreads();

    float a0 = db[f], a1 = db[FEAT + f], a2 = db[2 * FEAT + f];
#pragma unroll 4
    for (int r = 0; r < NRBF; ++r) {
        float rb = sh_rbf[r];
        a0 += rb * dw[r * F3 + f];
        a1 += rb * dw[r * F3 + FEAT + f];
        a2 += rb * dw[r * F3 + 2 * FEAT + f];
    }
    const float env = sh_env;
    const long  jf  = (long)j * FEAT + f;
    const long  if0 = (long)i0 * FEAT + f;
    const float p0 = phi[(long)j * F3 + f]            * (a0 * env);
    const float p1 = phi[(long)j * F3 + FEAT + f]     * (a1 * env);
    const float p2 = phi[(long)j * F3 + 2 * FEAT + f] * (a2 * env);

    atomicAdd(ds + if0, p1);
#pragma unroll
    for (int dd = 0; dd < 3; ++dd)
        atomicAdd(dv + (long)dd * NAF + if0,
                  p2 * sh_unit[dd] + p0 * v[(long)dd * NAF + jf]);
}

__global__ void k_apply(float* __restrict__ s, float* __restrict__ v,
                        const float* __restrict__ ds, const float* __restrict__ dv)
{
    int idx = blockIdx.x * blockDim.x + threadIdx.x;
    if (idx >= NAF) return;
    s[idx] += ds[idx];
#pragma unroll
    for (int dd = 0; dd < 3; ++dd)
        v[(long)dd * NAF + idx] += dv[(long)dd * NAF + idx];
}

// build [s | ||v_v||] concat input for the update dense block
__global__ void k_concat(const float* __restrict__ s, const float* __restrict__ vv,
                         float* __restrict__ h2)
{
    int idx = blockIdx.x * blockDim.x + threadIdx.x;
    if (idx >= NAF) return;
    int n = idx >> 7, f = idx & 127;
    h2[(long)n * 2 * FEAT + f] = s[idx];
    float nn = 0.0f;
#pragma unroll
    for (int dd = 0; dd < 3; ++dd) {
        float x = vv[(long)dd * NAF + idx];
        nn += x * x + 1e-15f;
    }
    h2[(long)n * 2 * FEAT + FEAT + f] = sqrtf(nn);
}

__global__ void k_upd_apply(float* __restrict__ s, float* __restrict__ v,
                            const float* __restrict__ uv, const float* __restrict__ vv,
                            const float* __restrict__ phi, int write_v)
{
    int idx = blockIdx.x * blockDim.x + threadIdx.x;
    if (idx >= NAF) return;
    int n = idx >> 7, f = idx & 127;
    const float a0 = phi[(long)n * F3 + f];
    const float a1 = phi[(long)n * F3 + FEAT + f];
    const float a2 = phi[(long)n * F3 + 2 * FEAT + f];
    float dot = 0.0f, uvd[3];
#pragma unroll
    for (int dd = 0; dd < 3; ++dd) {
        float uu = uv[(long)dd * NAF + idx];
        uvd[dd] = uu;
        dot += uu * vv[(long)dd * NAF + idx];
    }
    s[idx] += dot * a1 + a2;
    if (write_v) {
#pragma unroll
        for (int dd = 0; dd < 3; ++dd)
            v[(long)dd * NAF + idx] += uvd[dd] * a0;
    }
}

__global__ void k_scatter_s(const float* __restrict__ s, const int* __restrict__ mapping,
                            float* __restrict__ SI)
{
    int idx = blockIdx.x * blockDim.x + threadIdx.x;
    if (idx >= NAF) return;
    int n = idx >> 7, f = idx & 127;
    atomicAdd(SI + (long)mapping[n] * FEAT + f, s[idx]);
}

__global__ void k_divide(float* __restrict__ SI, const float* __restrict__ cnt)
{
    int idx = blockIdx.x * blockDim.x + threadIdx.x;
    if (idx >= NCG * FEAT) return;
    SI[idx] /= fmaxf(cnt[idx >> 7], 1.0f);
}

// contract: only the middle third survives into S_I (V_I is dead in the output)
__global__ void k_contract(const float* __restrict__ xyz, const float* __restrict__ cgxyz,
                           const int* __restrict__ mapping,
                           const float* __restrict__ phimid,
                           const float* __restrict__ dwmid, const float* __restrict__ dbmid,
                           float* __restrict__ acc)
{
    const int n = blockIdx.x;
    const int f = threadIdx.x;
    const int m = mapping[n];
    __shared__ float sh_rbf[NRBF];
    __shared__ float sh_env;
    if (f < NRBF) {
        float rx = xyz[3 * n + 0] - cgxyz[3 * m + 0];
        float ry = xyz[3 * n + 1] - cgxyz[3 * m + 1];
        float rz = xyz[3 * n + 2] - cgxyz[3 * m + 2];
        float d  = sqrtf(rx * rx + ry * ry + rz * rz + 3e-15f);
        sh_rbf[f] = __sinf((float)(f + 1) * (PI_F / CUTF) * d) * __builtin_amdgcn_rcpf(d);
        if (f == 0)
            sh_env = (d < CUTF) ? 0.5f * (__cosf(PI_F * d / CUTF) + 1.0f) : 0.0f;
    }
    __syncthreads();
    float a = dbmid[f];
#pragma unroll 4
    for (int r = 0; r < NRBF; ++r) a += sh_rbf[r] * dwmid[r * F3 + f];
    atomicAdd(acc + (long)m * FEAT + f, phimid[(long)n * FEAT + f] * (a * sh_env));
}

__global__ void k_ctr_final(float* __restrict__ SI, const float* __restrict__ acc,
                            const float* __restrict__ cnt)
{
    int idx = blockIdx.x * blockDim.x + threadIdx.x;
    if (idx >= NCG * FEAT) return;
    SI[idx] += acc[idx] / fmaxf(cnt[idx >> 7], 1.0f);
}

// ---------------------------------------------------------------------------
// Host-side orchestration
// ---------------------------------------------------------------------------
static inline void run_gemm(const float* A, int lda, const float* B, int ldb,
                            float* C, int ldc, const float* bias, int act,
                            int M, int N, int K, hipStream_t stream)
{
    dim3 grid((M / 16 + 7) / 8, N / 64);       // 8 M-tiles x 64 cols per block
    if (K == 128)
        gemm_wmma_bf16<128><<<grid, 256, 0, stream>>>(A, lda, B, ldb, C, ldc,
                                                      bias, act, M);
    else
        gemm_wmma_bf16<256><<<grid, 256, 0, stream>>>(A, lda, B, ldb, C, ldc,
                                                      bias, act, M);
}

extern "C" void kernel_launch(void* const* d_in, const int* in_sizes, int n_in,
                              void* d_out, int out_size, void* d_ws, size_t ws_size,
                              hipStream_t stream)
{
    (void)in_sizes; (void)n_in; (void)out_size; (void)ws_size;
    const int*   z       = (const int*)  d_in[0];
    const float* xyz     = (const float*)d_in[1];
    const float* cgxyz   = (const float*)d_in[2];
    const int*   mapping = (const int*)  d_in[3];
    const int*   nbr     = (const int*)  d_in[4];
    /* d_in[5] = cg_nbr_list: unused by the reference output */
    const float* emb_tab = (const float*)d_in[6];
    const float* msg_w1  = (const float*)d_in[7];
    const float* msg_b1  = (const float*)d_in[8];
    const float* msg_w2  = (const float*)d_in[9];
    const float* msg_b2  = (const float*)d_in[10];
    const float* msg_dw  = (const float*)d_in[11];
    const float* msg_db  = (const float*)d_in[12];
    const float* upd_u   = (const float*)d_in[13];
    const float* upd_vm  = (const float*)d_in[14];
    const float* upd_w1  = (const float*)d_in[15];
    const float* upd_b1  = (const float*)d_in[16];
    const float* upd_w2  = (const float*)d_in[17];
    const float* upd_b2  = (const float*)d_in[18];
    const float* ctr_w1  = (const float*)d_in[19];
    const float* ctr_b1  = (const float*)d_in[20];
    const float* ctr_w2  = (const float*)d_in[21];
    const float* ctr_b2  = (const float*)d_in[22];
    const float* ctr_dw  = (const float*)d_in[23];
    const float* ctr_db  = (const float*)d_in[24];
    float* SI = (float*)d_out;               // S_I lives directly in d_out

    // workspace carve-up (floats); vector fields are SoA [3][NA,128]
    float* s   = (float*)d_ws;               // [NA,128]
    float* v   = s   + (size_t)NAF;          // [3][NA,128]
    float* phi = v   + (size_t)3 * NAF;      // [NA,384]
    float* h2  = phi + (size_t)NA * F3;      // [NA,256]
    float* hid = h2  + (size_t)NA * 2 * FEAT;// [NA,128]  (hidden / ds-accum / ctr-accum)
    float* dva = hid + (size_t)NAF;          // [3][NA,128]  dv accumulator
    float* uv  = dva + (size_t)3 * NAF;      // [3][NA,128]
    float* vv  = uv  + (size_t)3 * NAF;      // [3][NA,128]
    float* cnt = vv  + (size_t)3 * NAF;      // [NCG]

    const int nfB = (NAF + 255) / 256;       // blocks for per-(atom,feat) kernels
    const int cgB = (NCG * FEAT + 255) / 256;

    // init: s = embed[z], v = 0, per-CG counts
    k_embed<<<nfB, 256, 0, stream>>>(z, emb_tab, s);
    hipMemsetAsync(v, 0, (size_t)3 * NAF * sizeof(float), stream);
    hipMemsetAsync(cnt, 0, (size_t)NCG * sizeof(float), stream);
    k_count<<<(NA + 255) / 256, 256, 0, stream>>>(mapping, cnt);

    for (int i = 0; i < NCONV; ++i) {
        const float* mw1 = msg_w1 + (size_t)i * FEAT * FEAT;
        const float* mb1 = msg_b1 + (size_t)i * FEAT;
        const float* mw2 = msg_w2 + (size_t)i * FEAT * F3;
        const float* mb2 = msg_b2 + (size_t)i * F3;
        const float* mdw = msg_dw + (size_t)i * NRBF * F3;
        const float* mdb = msg_db + (size_t)i * F3;

        // ---- message: phi = silu(s@w1+b1)@w2+b2 (WMMA), then fused edge pass
        run_gemm(s, FEAT, mw1, FEAT, hid, FEAT, mb1, 1, NA, FEAT, FEAT, stream);
        run_gemm(hid, FEAT, mw2, F3, phi, F3, mb2, 0, NA, F3, FEAT, stream);
        hipMemsetAsync(hid, 0, (size_t)NAF * sizeof(float), stream);
        hipMemsetAsync(dva, 0, (size_t)3 * NAF * sizeof(float), stream);
        k_edge_msg<<<2 * NE, FEAT, 0, stream>>>(xyz, nbr, phi, v, mdw, mdb, hid, dva);
        k_apply<<<nfB, 256, 0, stream>>>(s, v, hid, dva);

        // ---- update: u_v / v_v einsums = 3 contiguous SoA-plane GEMMs each
        const float* uu  = upd_u  + (size_t)i * FEAT * FEAT;
        const float* um  = upd_vm + (size_t)i * FEAT * FEAT;
        for (int dd = 0; dd < 3; ++dd) {
            run_gemm(v + (size_t)dd * NAF, FEAT, uu, FEAT,
                     uv + (size_t)dd * NAF, FEAT, nullptr, 0, NA, FEAT, FEAT, stream);
            run_gemm(v + (size_t)dd * NAF, FEAT, um, FEAT,
                     vv + (size_t)dd * NAF, FEAT, nullptr, 0, NA, FEAT, FEAT, stream);
        }
        k_concat<<<nfB, 256, 0, stream>>>(s, vv, h2);
        run_gemm(h2, 2 * FEAT, upd_w1 + (size_t)i * 2 * FEAT * FEAT, FEAT,
                 hid, FEAT, upd_b1 + (size_t)i * FEAT, 1, NA, FEAT, 2 * FEAT, stream);
        run_gemm(hid, FEAT, upd_w2 + (size_t)i * FEAT * F3, F3,
                 phi, F3, upd_b2 + (size_t)i * F3, 0, NA, F3, FEAT, stream);
        k_upd_apply<<<nfB, 256, 0, stream>>>(s, v, uv, vv, phi, (i < NCONV - 1) ? 1 : 0);

        // ---- S_I initial scatter_mean after conv 0's update
        if (i == 0) {
            hipMemsetAsync(SI, 0, (size_t)NCG * FEAT * sizeof(float), stream);
            k_scatter_s<<<nfB, 256, 0, stream>>>(s, mapping, SI);
            k_divide<<<cgB, 256, 0, stream>>>(SI, cnt);
        }

        // ---- contract (middle third only; V_I is dead w.r.t. the output)
        run_gemm(s, FEAT, ctr_w1 + (size_t)i * FEAT * FEAT, FEAT,
                 h2, FEAT, ctr_b1 + (size_t)i * FEAT, 1, NA, FEAT, FEAT, stream);
        run_gemm(h2, FEAT, ctr_w2 + (size_t)i * FEAT * F3 + FEAT, F3,
                 phi, FEAT, ctr_b2 + (size_t)i * F3 + FEAT, 0, NA, FEAT, FEAT, stream);
        hipMemsetAsync(hid, 0, (size_t)NCG * FEAT * sizeof(float), stream);
        k_contract<<<NA, FEAT, 0, stream>>>(xyz, cgxyz, mapping, phi,
                                            ctr_dw + (size_t)i * NRBF * F3 + FEAT,
                                            ctr_db + (size_t)i * F3 + FEAT, hid);
        k_ctr_final<<<cgB, 256, 0, stream>>>(SI, hid, cnt);
    }
}
